// FirstStageRetriever_37864431682210
// MI455X (gfx1250) — matrix operand order repaired and checked by reading, not compiled, not run on hardware
//
#include <hip/hip_runtime.h>
#include <math.h>

// ---------------------------------------------------------------------------
// FirstStageRetriever for MI455X (gfx1250, wave32, WMMA)
//   sims       = normalize(q) @ normalize(c)^T           (WMMA f16->f32)
//   soft_sort  = w - PAVA(w - sort_desc(sims_row)),  w = [n..1]/eps
//   soft_rank  = scatter(s - PAVA(s - [n..1])),      s = theta sorted desc
// ---------------------------------------------------------------------------

#define INV_EPS 10.0f   // 1 / REG_STRENGTH (0.1)

typedef __attribute__((ext_vector_type(16))) _Float16 v16h;
typedef __attribute__((ext_vector_type(8)))  float    v8f;

// ---------------- row inverse norms (one wave per row) ----------------------
__global__ void rownorm_kernel(const float* __restrict__ x,
                               float* __restrict__ inv,
                               int rows, int K) {
  int wave = threadIdx.x >> 5;
  int lane = threadIdx.x & 31;
  int row  = blockIdx.x * (blockDim.x >> 5) + wave;
  if (row >= rows) return;
  const float* p = x + (size_t)row * K;
  float s = 0.f;
  for (int k = lane; k < K; k += 32) { float v = p[k]; s += v * v; }
#pragma unroll
  for (int off = 16; off >= 1; off >>= 1) s += __shfl_xor(s, off, 32);
  if (lane == 0) inv[row] = 1.0f / fmaxf(sqrtf(s), 1e-12f);
}

// ---------------- WMMA GEMM: sims = q @ c^T, scaled by invq*invc ------------
// One 16x16 output tile per wave. K=1024 consumed in steps of 32 via
// v_wmma_f32_16x16x32_f16. f32 inputs converted to f16 in registers.
// Corpus (64 MB f32) is read once and is fully L2-resident (192 MB L2),
// so the second m-tile's reads hit in L2: HBM traffic ~= one corpus pass.
__global__ void __launch_bounds__(256)
wmma_gemm_kernel(const float* __restrict__ q,      // [B x K]
                 const float* __restrict__ c,      // [N x K]
                 const float* __restrict__ invq,   // [B]
                 const float* __restrict__ invc,   // [N]
                 float* __restrict__ sims,         // [B x N]
                 int B, int N, int K) {
  const int m_tiles = B >> 4;                       // 2
  int wave = threadIdx.x >> 5;
  int lane = threadIdx.x & 31;
  int tile = blockIdx.x * (blockDim.x >> 5) + wave; // 0 .. (B/16)*(N/16)-1
  int tile_m = tile % m_tiles;
  int tile_n = tile / m_tiles;
  if (tile_n >= (N >> 4)) return;

  int half = lane >> 4;      // 0: lanes 0-15, 1: lanes 16-31
  int l16  = lane & 15;
  int m = tile_m * 16 + l16; // A: row this lane supplies
  int n = tile_n * 16 + l16; // B: corpus row (column of B) this lane supplies
  const float* qrow = q + (size_t)m * K;
  const float* crow = c + (size_t)n * K;
  const int akb = half * 8;   // A K sub-offset within each 16-chunk
  const int bkb = half * 8;   // B K sub-offset within each 16-chunk

  v8f acc = {};
#pragma unroll 2
  for (int k0 = 0; k0 < K; k0 += 32) {
    if (k0 + 32 < K) __builtin_prefetch(crow + k0 + 32 + bkb, 0, 3);

    // ---- A fragment: 16x32 f16. halves 0..7 = K[k0+akb..+7],
    //                  halves 8..15 = K[k0+16+akb..+7]
    float4 a0 = *(const float4*)(qrow + k0 + akb);
    float4 a1 = *(const float4*)(qrow + k0 + akb + 4);
    float4 a2 = *(const float4*)(qrow + k0 + 16 + akb);
    float4 a3 = *(const float4*)(qrow + k0 + 16 + akb + 4);
    v16h a;
    a[0]=(_Float16)a0.x; a[1]=(_Float16)a0.y; a[2]=(_Float16)a0.z; a[3]=(_Float16)a0.w;
    a[4]=(_Float16)a1.x; a[5]=(_Float16)a1.y; a[6]=(_Float16)a1.z; a[7]=(_Float16)a1.w;
    a[8]=(_Float16)a2.x; a[9]=(_Float16)a2.y; a[10]=(_Float16)a2.z; a[11]=(_Float16)a2.w;
    a[12]=(_Float16)a3.x; a[13]=(_Float16)a3.y; a[14]=(_Float16)a3.z; a[15]=(_Float16)a3.w;

    // ---- B fragment: 32x16 f16. halves 0..7 = K[k0+bkb..+7], halves 8..15 = K[k0+16+bkb..+7]
    float4 b0 = *(const float4*)(crow + k0 + bkb);
    float4 b1 = *(const float4*)(crow + k0 + bkb + 4);
    float4 b2 = *(const float4*)(crow + k0 + 16 + bkb);
    float4 b3 = *(const float4*)(crow + k0 + 16 + bkb + 4);
    v16h b;
    b[0]=(_Float16)b0.x; b[1]=(_Float16)b0.y; b[2]=(_Float16)b0.z; b[3]=(_Float16)b0.w;
    b[4]=(_Float16)b1.x; b[5]=(_Float16)b1.y; b[6]=(_Float16)b1.z; b[7]=(_Float16)b1.w;
    b[8]=(_Float16)b2.x; b[9]=(_Float16)b2.y; b[10]=(_Float16)b2.z; b[11]=(_Float16)b2.w;
    b[12]=(_Float16)b3.x; b[13]=(_Float16)b3.y; b[14]=(_Float16)b3.z; b[15]=(_Float16)b3.w;

    acc = __builtin_amdgcn_wmma_f32_16x16x32_f16(
        /*neg_a=*/false, a, /*neg_b=*/false, b,
        /*c_mod=*/(short)0, acc, /*reuse_a=*/false, /*reuse_b=*/false);
  }

  // D layout: VGPR r -> M = r + 8*half, N = l16
  float ic = invc[n];
#pragma unroll
  for (int r = 0; r < 8; ++r) {
    int gm = tile_m * 16 + r + 8 * half;
    sims[(size_t)gm * N + tile_n * 16 + l16] = acc[r] * invq[gm] * ic;
  }
}

// ---------------- per-row descending bitonic key/index sort in LDS ----------
// One block (1024 threads) per row. CDNA5 allows up to 320KB LDS per
// workgroup; one row needs 16384*(4+4) = 128KB, passed as dynamic LDS.
__global__ void __launch_bounds__(1024)
bitonic_kernel(const float* __restrict__ sims,
               float* __restrict__ keys, int* __restrict__ idx, int n) {
  extern __shared__ char smem[];
  float* kk = (float*)smem;            // [n]
  int*   ii = (int*)(smem + (size_t)n * sizeof(float)); // [n]

  int row = blockIdx.x;
  const float* src = sims + (size_t)row * n;
  for (int i = threadIdx.x; i < n; i += blockDim.x) { kk[i] = src[i]; ii[i] = i; }
  __syncthreads();

  for (int k = 2; k <= n; k <<= 1) {
    for (int j = k >> 1; j > 0; j >>= 1) {
      for (int i = threadIdx.x; i < n; i += blockDim.x) {
        int l = i ^ j;
        if (l > i) {
          float ki = kk[i], kl = kk[l];
          bool upSeg = ((i & k) == 0);
          // descending overall: invert classic ascending comparators
          bool sw = upSeg ? (ki < kl) : (ki > kl);
          if (sw) {
            kk[i] = kl; kk[l] = ki;
            int t = ii[i]; ii[i] = ii[l]; ii[l] = t;
          }
        }
      }
      __syncthreads();
    }
  }

  float* ko = keys + (size_t)row * n;
  int*   io = idx  + (size_t)row * n;
  for (int i = threadIdx.x; i < n; i += blockDim.x) { ko[i] = kk[i]; io[i] = ii[i]; }
}

// ---------------- build PAVA inputs -----------------------------------------
// y[task]    task in [0,B)   : y1 = w - s,        w = (n-i)*INV_EPS  (soft sort)
// y[task+B]  task in [0,B)   : y2 = s2 - (n-i),   s2 = -keys[n-1-i]*INV_EPS (rank)
__global__ void prep_kernel(const float* __restrict__ keys,
                            float* __restrict__ y, int B, int n) {
  int row = blockIdx.y;
  int i = blockIdx.x * blockDim.x + threadIdx.x;
  if (i >= n) return;
  const float* s1 = keys + (size_t)row * n;
  float w = (float)(n - i);
  y[(size_t)row * n + i] = w * INV_EPS - s1[i];
  float s2 = -s1[n - 1 - i] * INV_EPS;
  y[(size_t)(row + B) * n + i] = s2 - w;
}

// ---------------- PAVA (pool adjacent violators), in place ------------------
// One thread per task (2*B tasks). Stack in workspace, solution overwrites y.
__global__ void pava_kernel(float* __restrict__ y,
                            float* __restrict__ sums,
                            float* __restrict__ cnts,
                            int tasks, int n) {
  int task = blockIdx.x * blockDim.x + threadIdx.x;
  if (task >= tasks) return;
  float* yy = y    + (size_t)task * n;
  float* s  = sums + (size_t)task * n;
  float* c  = cnts + (size_t)task * n;
  int top = 0;
  for (int i = 0; i < n; ++i) {
    float sv = yy[i], cv = 1.0f;
    while (top > 0 && (s[top - 1] / c[top - 1]) <= (sv / cv)) {
      sv += s[top - 1];
      cv += c[top - 1];
      --top;
    }
    s[top] = sv; c[top] = cv; ++top;
  }
  int pos = 0;
  for (int b = 0; b < top; ++b) {
    float mean = s[b] / c[b];
    int cnt = (int)c[b];
    for (int t = 0; t < cnt; ++t) yy[pos++] = mean;
  }
}

// ---------------- assemble outputs ------------------------------------------
__global__ void final_kernel(const float* __restrict__ y,     // PAVA solutions
                             const float* __restrict__ keys,
                             const int* __restrict__ idx,
                             float* __restrict__ out_sorted,
                             float* __restrict__ out_ranks,
                             int B, int n) {
  int row = blockIdx.y;
  int i = blockIdx.x * blockDim.x + threadIdx.x;
  if (i >= n) return;
  float w = (float)(n - i) * INV_EPS;
  out_sorted[(size_t)row * n + i] = w - y[(size_t)row * n + i];
  // soft rank: s2 descending-theta order; scatter through reversed perm
  float s2 = -keys[(size_t)row * n + (n - 1 - i)] * INV_EPS;
  float primal = s2 - y[(size_t)(row + B) * n + i];
  int dest = idx[(size_t)row * n + (n - 1 - i)];
  out_ranks[(size_t)row * n + dest] = primal;
}

// ---------------------------------------------------------------------------
extern "C" void kernel_launch(void* const* d_in, const int* in_sizes, int n_in,
                              void* d_out, int out_size, void* d_ws, size_t ws_size,
                              hipStream_t stream) {
  const int K = 1024;
  const int B = in_sizes[0] / K;   // 32
  const int N = in_sizes[1] / K;   // 16384

  const float* q = (const float*)d_in[0];
  const float* c = (const float*)d_in[1];

  float* out      = (float*)d_out;
  float* sims     = out;                       // [B x N]
  float* o_sorted = out + (size_t)B * N;       // [B x N]
  float* o_ranks  = out + (size_t)2 * B * N;   // [B x N]

  // workspace layout (floats)
  float* ws   = (float*)d_ws;
  float* invq = ws;                      size_t off = 64;
  float* invc = ws + off;                off += (size_t)N;
  float* keys = ws + off;                off += (size_t)B * N;
  int*   idx  = (int*)(ws + off);        off += (size_t)B * N;
  float* y    = ws + off;                off += (size_t)2 * B * N;  // y1|y2
  float* sums = ws + off;                off += (size_t)2 * B * N;
  float* cnts = ws + off;                off += (size_t)2 * B * N;
  (void)ws_size; (void)n_in; (void)out_size;

  // 1) inverse row norms
  rownorm_kernel<<<dim3((B + 7) / 8), dim3(256), 0, stream>>>(q, invq, B, K);
  rownorm_kernel<<<dim3((N + 7) / 8), dim3(256), 0, stream>>>(c, invc, N, K);

  // 2) WMMA GEMM -> sims
  int tiles = (B / 16) * (N / 16);             // 2048
  wmma_gemm_kernel<<<dim3(tiles / 8), dim3(256), 0, stream>>>(
      q, c, invq, invc, sims, B, N, K);

  // 3) per-row descending sort (values + indices) in LDS (128KB dynamic)
  size_t lds_bytes = (size_t)N * (sizeof(float) + sizeof(int));
  bitonic_kernel<<<dim3(B), dim3(1024), lds_bytes, stream>>>(sims, keys, idx, N);

  // 4) PAVA inputs
  prep_kernel<<<dim3(N / 256, B), dim3(256), 0, stream>>>(keys, y, B, N);

  // 5) PAVA (2*B independent sequential tasks)
  pava_kernel<<<dim3(1), dim3(2 * B), 0, stream>>>(y, sums, cnts, 2 * B, N);

  // 6) final soft-sort / soft-rank outputs
  final_kernel<<<dim3(N / 256, B), dim3(256), 0, stream>>>(
      y, keys, idx, o_sorted, o_ranks, B, N);
}
